// NPFiLM_Decoder_84834194030793
// MI455X (gfx1250) — matrix operand order repaired and checked by reading, not compiled, not run on hardware
//
#include <hip/hip_runtime.h>
#include <hip/hip_bf16.h>

typedef __attribute__((ext_vector_type(16))) _Float16 v16h;
typedef __attribute__((ext_vector_type(8)))  _Float16 v8h;
typedef __attribute__((ext_vector_type(4)))  _Float16 v4h;
typedef __attribute__((ext_vector_type(8)))  float    v8f;

#define N_TOK   8192
#define P_DIM   16
#define INDIM   256
#define OUTDIM  128
#define H0      512
#define H1      1024
#define H2      1024
#define H3      512
#define BM      64          // rows (tokens) per block
#define NWAVES  8           // 256 threads / wave32

// ---------------------------------------------------------------------------
// f32 -> f16 weight conversion (once per launch; weights then live in L2)
// ---------------------------------------------------------------------------
__global__ void npfilm_cvt_f16(const float* __restrict__ src,
                               _Float16* __restrict__ dst, int n) {
    int stride = gridDim.x * blockDim.x;
    for (int i = blockIdx.x * blockDim.x + threadIdx.x; i < n; i += stride)
        dst[i] = (_Float16)src[i];
}

// ---------------------------------------------------------------------------
// One GEMM stage: out[BM x No] = relu( in[BM x K] * W[No x K]^T + bias )
// A from LDS, B streamed from global (f16), C accumulated in f32 WMMA tiles.
// Per 16-wide output strip a wave keeps 4 row-tiles live so every B fragment
// is loaded once and reused by 4 v_wmma_f32_16x16x32_f16 ops.
// ---------------------------------------------------------------------------
__device__ __forceinline__ void gemm_stage(
    const _Float16* __restrict__ in, int K,
    const _Float16* __restrict__ Wg, const float* __restrict__ bias,
    _Float16* __restrict__ outl, int No, bool relu,
    int wave, int ln15, int lhi)
{
    const int nstrips = No >> 4;
    for (int s = wave; s < nstrips; s += NWAVES) {
        const int n = (s << 4) + ln15;                       // output column (lane)
        const _Float16* wrow = Wg + (size_t)n * K + (lhi << 3);
        v8f acc0 = {}, acc1 = {}, acc2 = {}, acc3 = {};
        const _Float16* abase = in + (size_t)ln15 * K + (lhi << 3);
        for (int k0 = 0; k0 < K; k0 += 32) {
            // B fragment: lane holds column n; same K layout as A:
            // lanes 0-15 K=k0..+7,k0+16..+23; lanes 16-31 K=k0+8..+15,k0+24..+31.
            v8h blo = *(const v8h*)(wrow + k0);
            v8h bhi = *(const v8h*)(wrow + k0 + 16);
            v16h b = __builtin_shufflevector(blo, bhi, 0,1,2,3,4,5,6,7,8,9,10,11,12,13,14,15);
            // A fragments, rows m = tm*16 + ln15. CDNA5 16-bit A layout:
            // lanes 0-15: elems0-7=K k0..+7, elems8-15=K k0+16..+23
            // lanes16-31: elems0-7=K k0+8..+15, elems8-15=K k0+24..+31
            const _Float16* ar0 = abase + k0;
            v8h lo0 = *(const v8h*)(ar0);
            v8h hi0 = *(const v8h*)(ar0 + 16);
            v16h a0 = __builtin_shufflevector(lo0, hi0, 0,1,2,3,4,5,6,7,8,9,10,11,12,13,14,15);
            acc0 = __builtin_amdgcn_wmma_f32_16x16x32_f16(false, a0, false, b, (short)0, acc0, false, false);

            const _Float16* ar1 = ar0 + (size_t)16 * K;
            v8h lo1 = *(const v8h*)(ar1);
            v8h hi1 = *(const v8h*)(ar1 + 16);
            v16h a1 = __builtin_shufflevector(lo1, hi1, 0,1,2,3,4,5,6,7,8,9,10,11,12,13,14,15);
            acc1 = __builtin_amdgcn_wmma_f32_16x16x32_f16(false, a1, false, b, (short)0, acc1, false, false);

            const _Float16* ar2 = ar1 + (size_t)16 * K;
            v8h lo2 = *(const v8h*)(ar2);
            v8h hi2 = *(const v8h*)(ar2 + 16);
            v16h a2 = __builtin_shufflevector(lo2, hi2, 0,1,2,3,4,5,6,7,8,9,10,11,12,13,14,15);
            acc2 = __builtin_amdgcn_wmma_f32_16x16x32_f16(false, a2, false, b, (short)0, acc2, false, false);

            const _Float16* ar3 = ar2 + (size_t)16 * K;
            v8h lo3 = *(const v8h*)(ar3);
            v8h hi3 = *(const v8h*)(ar3 + 16);
            v16h a3 = __builtin_shufflevector(lo3, hi3, 0,1,2,3,4,5,6,7,8,9,10,11,12,13,14,15);
            acc3 = __builtin_amdgcn_wmma_f32_16x16x32_f16(false, a3, false, b, (short)0, acc3, false, false);
        }
        // Epilogue: C/D layout -> elem i is row (tm*16 + lhi*8 + i), col n.
        const float bv = bias[n];
#define NPFILM_EPI(ACC, TM)                                                    \
        {                                                                      \
            _Float16* o = outl + ((size_t)((TM)*16 + lhi*8)) * No + n;         \
            _Pragma("unroll")                                                  \
            for (int i = 0; i < 8; ++i) {                                      \
                float v = (ACC)[i] + bv;                                       \
                if (relu) v = fmaxf(v, 0.0f);                                  \
                o[(size_t)i * No] = (_Float16)v;                               \
            }                                                                  \
        }
        NPFILM_EPI(acc0, 0) NPFILM_EPI(acc1, 1) NPFILM_EPI(acc2, 2) NPFILM_EPI(acc3, 3)
#undef NPFILM_EPI
    }
}

// ---------------------------------------------------------------------------
// Fully fused MLP: x -> L_in -> L1 -> L2 -> L3 -> L_out -> (mu, var)
// ---------------------------------------------------------------------------
__global__ __launch_bounds__(256, 1)
void npfilm_fused(const float* __restrict__ x, const unsigned char* __restrict__ mask,
                  const _Float16* __restrict__ wWin, const _Float16* __restrict__ wW1,
                  const _Float16* __restrict__ wW2,  const _Float16* __restrict__ wW3,
                  const _Float16* __restrict__ wWout,
                  const float* __restrict__ b_in, const float* __restrict__ b1,
                  const float* __restrict__ b2,   const float* __restrict__ b3,
                  const float* __restrict__ b_out,
                  float* __restrict__ out)
{
    const int tid  = threadIdx.x;
    const int wave = tid >> 5;
    const int lane = tid & 31;
    const int ln15 = lane & 15;
    const int lhi  = lane >> 4;
    const int p    = blockIdx.x & (P_DIM - 1);
    const int n0   = (blockIdx.x / P_DIM) * BM;

    extern __shared__ _Float16 lds[];
    _Float16* bufA = lds;                      // up to BM x 1024
    _Float16* bufB = lds + (size_t)BM * 1024;  // up to BM x 1024

    // ---- stage in: x tile (BM x 256 f32) -> bufA as f16 ----
    for (int idx = tid; idx < BM * INDIM / 4; idx += 256) {
        const int r  = idx >> 6;            // 64 float4 per row
        const int c4 = idx & 63;
        const float4 v = ((const float4*)(x + ((size_t)(n0 + r) * P_DIM + p) * INDIM))[c4];
        v4h h;
        h[0] = (_Float16)v.x; h[1] = (_Float16)v.y;
        h[2] = (_Float16)v.z; h[3] = (_Float16)v.w;
        *(v4h*)(bufA + (size_t)r * INDIM + c4 * 4) = h;
    }
    __syncthreads();

    // ---- L_in (per-p): 256 -> 512, no relu ----
    gemm_stage(bufA, INDIM, wWin + (size_t)p * H0 * INDIM, b_in + (size_t)p * H0,
               bufB, H0, false, wave, ln15, lhi);
    __syncthreads();
    // ---- L1: 512 -> 1024, relu ----
    gemm_stage(bufB, H0, wW1, b1, bufA, H1, true, wave, ln15, lhi);
    __syncthreads();
    // ---- L2: 1024 -> 1024, relu ----
    gemm_stage(bufA, H1, wW2, b2, bufB, H2, true, wave, ln15, lhi);
    __syncthreads();
    // ---- L3: 1024 -> 512, relu ----
    gemm_stage(bufB, H2, wW3, b3, bufA, H3, true, wave, ln15, lhi);
    __syncthreads();

    // ---- L_out (per-p): 512 -> 256, then mu / softplus-var head ----
    const _Float16* Wo = wWout + (size_t)p * (2 * OUTDIM) * H3;
    const float*    bo = b_out + (size_t)p * (2 * OUTDIM);
    const float NANF = __builtin_nanf("");
    const size_t var_base = (size_t)N_TOK * P_DIM * OUTDIM;

    for (int s = wave; s < (2 * OUTDIM) / 16; s += NWAVES) {
        const int n = (s << 4) + ln15;
        const _Float16* wrow = Wo + (size_t)n * H3 + (lhi << 3);
        v8f acc0 = {}, acc1 = {}, acc2 = {}, acc3 = {};
        const _Float16* abase = bufA + (size_t)ln15 * H3 + (lhi << 3);
        for (int k0 = 0; k0 < H3; k0 += 32) {
            v8h blo = *(const v8h*)(wrow + k0);
            v8h bhi = *(const v8h*)(wrow + k0 + 16);
            v16h b = __builtin_shufflevector(blo, bhi, 0,1,2,3,4,5,6,7,8,9,10,11,12,13,14,15);
            const _Float16* ar0 = abase + k0;
            v8h lo0 = *(const v8h*)(ar0);
            v8h hi0 = *(const v8h*)(ar0 + 16);
            v16h a0 = __builtin_shufflevector(lo0, hi0, 0,1,2,3,4,5,6,7,8,9,10,11,12,13,14,15);
            acc0 = __builtin_amdgcn_wmma_f32_16x16x32_f16(false, a0, false, b, (short)0, acc0, false, false);
            const _Float16* ar1 = ar0 + (size_t)16 * H3;
            v8h lo1 = *(const v8h*)(ar1);
            v8h hi1 = *(const v8h*)(ar1 + 16);
            v16h a1 = __builtin_shufflevector(lo1, hi1, 0,1,2,3,4,5,6,7,8,9,10,11,12,13,14,15);
            acc1 = __builtin_amdgcn_wmma_f32_16x16x32_f16(false, a1, false, b, (short)0, acc1, false, false);
            const _Float16* ar2 = ar1 + (size_t)16 * H3;
            v8h lo2 = *(const v8h*)(ar2);
            v8h hi2 = *(const v8h*)(ar2 + 16);
            v16h a2 = __builtin_shufflevector(lo2, hi2, 0,1,2,3,4,5,6,7,8,9,10,11,12,13,14,15);
            acc2 = __builtin_amdgcn_wmma_f32_16x16x32_f16(false, a2, false, b, (short)0, acc2, false, false);
            const _Float16* ar3 = ar2 + (size_t)16 * H3;
            v8h lo3 = *(const v8h*)(ar3);
            v8h hi3 = *(const v8h*)(ar3 + 16);
            v16h a3 = __builtin_shufflevector(lo3, hi3, 0,1,2,3,4,5,6,7,8,9,10,11,12,13,14,15);
            acc3 = __builtin_amdgcn_wmma_f32_16x16x32_f16(false, a3, false, b, (short)0, acc3, false, false);
        }
        const float bv = bo[n];
#define NPFILM_OUT(ACC, TM)                                                    \
        {                                                                      \
            _Pragma("unroll")                                                  \
            for (int i = 0; i < 8; ++i) {                                      \
                const int r = (TM) * 16 + lhi * 8 + i;                         \
                const size_t t = (size_t)(n0 + r) * P_DIM + p;                 \
                const bool msk = mask[t] != 0;                                 \
                float v = (ACC)[i] + bv;                                       \
                if (n < OUTDIM) {                                              \
                    out[t * OUTDIM + n] = msk ? NANF : v;                      \
                } else {                                                       \
                    float sp  = fmaxf(v, 0.0f) + log1pf(expf(-fabsf(v)));      \
                    float var = 0.01f + 0.99f * sp;                            \
                    out[var_base + t * OUTDIM + (n - OUTDIM)] = msk ? NANF : var; \
                }                                                              \
            }                                                                  \
        }
        NPFILM_OUT(acc0, 0) NPFILM_OUT(acc1, 1) NPFILM_OUT(acc2, 2) NPFILM_OUT(acc3, 3)
#undef NPFILM_OUT
    }
}

// ---------------------------------------------------------------------------
extern "C" void kernel_launch(void* const* d_in, const int* in_sizes, int n_in,
                              void* d_out, int out_size, void* d_ws, size_t ws_size,
                              hipStream_t stream) {
    (void)in_sizes; (void)n_in; (void)out_size; (void)ws_size;
    const float*         x     = (const float*)d_in[0];
    const unsigned char* mask  = (const unsigned char*)d_in[1];
    const float*         W_in  = (const float*)d_in[2];
    const float*         b_in  = (const float*)d_in[3];
    const float*         W1    = (const float*)d_in[4];
    const float*         b1    = (const float*)d_in[5];
    const float*         W2    = (const float*)d_in[6];
    const float*         b2    = (const float*)d_in[7];
    const float*         W3    = (const float*)d_in[8];
    const float*         b3    = (const float*)d_in[9];
    const float*         W_out = (const float*)d_in[10];
    const float*         b_out = (const float*)d_in[11];

    // f16 weight copies in workspace (~12.6 MB)
    _Float16* w16   = (_Float16*)d_ws;
    _Float16* wWin  = w16;
    _Float16* wW1   = wWin + (size_t)P_DIM * H0 * INDIM;
    _Float16* wW2   = wW1  + (size_t)H1 * H0;
    _Float16* wW3   = wW2  + (size_t)H2 * H1;
    _Float16* wWout = wW3  + (size_t)H3 * H2;

    npfilm_cvt_f16<<<512, 256, 0, stream>>>(W_in,  wWin,  P_DIM * H0 * INDIM);
    npfilm_cvt_f16<<<512, 256, 0, stream>>>(W1,    wW1,   H1 * H0);
    npfilm_cvt_f16<<<512, 256, 0, stream>>>(W2,    wW2,   H2 * H1);
    npfilm_cvt_f16<<<512, 256, 0, stream>>>(W3,    wW3,   H3 * H2);
    npfilm_cvt_f16<<<512, 256, 0, stream>>>(W_out, wWout, P_DIM * 2 * OUTDIM * H3);

    const size_t shmem = (size_t)2 * BM * 1024 * sizeof(_Float16); // 256 KB
    hipFuncSetAttribute((const void*)npfilm_fused,
                        hipFuncAttributeMaxDynamicSharedMemorySize, (int)shmem);
    npfilm_fused<<<dim3(P_DIM * (N_TOK / BM)), 256, shmem, stream>>>(
        x, mask, wWin, wW1, wW2, wW3, wWout,
        b_in, b1, b2, b3, b_out, (float*)d_out);
}
